// MRFLoss_68178310857069
// MI455X (gfx1250) — compile-verified, matrix operand order checked by reading, not run on hardware
//
#include <hip/hip_runtime.h>
#include <math.h>

typedef __attribute__((ext_vector_type(16))) _Float16 v16h;
typedef __attribute__((ext_vector_type(8)))  _Float16 v8h;
typedef __attribute__((ext_vector_type(8)))  float    v8f;

#define NCLS      1000
#define NCLS_PAD  1024
#define DF        512
#define EMA_D     0.999f
#define LAMBDA_R  0.1f
#define AS        520   // LDS row stride in halves (512 + 8 pad -> bank spread)

// ---------------- workspace layout (float units) ----------------
// [0      , 4096 )  sq[b] = ||features_b||^2            (float)
// [4096   , +1024)  counts  (int)
// [5120   , +1024)  cursor  (int)
// [6144   , +1024)  offsets (int)
// [7168   , +4096)  member  (int)
// [11264  , +1024)  s_arr   (float)
// [12288  , +1024)  var_fin (float)
// [13312  , +8   )  scalars: 0=sum_s 1=bar_s 2=ce_acc 3=rep_acc

__global__ void zero_kernel(int* counts, int* cursor, float* scalars) {
    int i = blockIdx.x * blockDim.x + threadIdx.x;
    if (i < NCLS_PAD) { counts[i] = 0; cursor[i] = 0; }
    if (i < 8) scalars[i] = 0.0f;
}

__global__ void hist_kernel(const int* __restrict__ targets, int* counts, int n) {
    int i = blockIdx.x * blockDim.x + threadIdx.x;
    if (i < n) atomicAdd(&counts[targets[i]], 1);
}

__global__ __launch_bounds__(1024) void scan_kernel(const int* counts, int* offsets) {
    __shared__ int tmp[NCLS_PAD];
    int tid = threadIdx.x;
    int v0 = counts[tid];
    tmp[tid] = v0; __syncthreads();
    for (int o = 1; o < NCLS_PAD; o <<= 1) {
        int v = (tid >= o) ? tmp[tid - o] : 0;
        __syncthreads();
        tmp[tid] += v;
        __syncthreads();
    }
    offsets[tid] = tmp[tid] - v0;   // exclusive scan
}

__global__ void scatter_kernel(const int* __restrict__ targets, const int* __restrict__ offsets,
                               int* cursor, int* member, int n) {
    int i = blockIdx.x * blockDim.x + threadIdx.x;
    if (i < n) {
        int t = targets[i];
        int p = atomicAdd(&cursor[t], 1);
        member[offsets[t] + p] = i;
    }
}

// one wave per row: sq[b] = sum f^2 (exact f32)
__global__ __launch_bounds__(256) void sq_kernel(const float* __restrict__ features,
                                                 float* __restrict__ sq, int n) {
    int w = threadIdx.x >> 5, lane = threadIdx.x & 31;
    int b = blockIdx.x * 8 + w;
    if (b >= n) return;
    const float* f = features + (size_t)b * DF;
    float acc = 0.0f;
    for (int e = lane; e < DF; e += 32) { float x = f[e]; acc += x * x; }
    #pragma unroll
    for (int o = 16; o >= 1; o >>= 1) acc += __shfl_down(acc, o, 32);
    if (lane == 0) sq[b] = acc;
}

// per-class: mean, ||mean||_5^2, variance, EMA update, s = cbrt(mn+var)
__global__ __launch_bounds__(256) void cls_stats_kernel(
    const float* __restrict__ features, const float* __restrict__ mn_in_arr,
    const float* __restrict__ var_in_arr, const int* __restrict__ counts,
    const int* __restrict__ offsets, const int* __restrict__ member,
    float* __restrict__ s_arr, float* __restrict__ var_fin) {
    __shared__ float meanb[DF];
    __shared__ float red[256];
    int c = blockIdx.x, tid = threadIdx.x;
    int m = counts[c];
    float mn_in = mn_in_arr[c], var_in = var_in_arr[c];
    if (m == 0) {
        if (tid == 0) { var_fin[c] = var_in; s_arr[c] = cbrtf(fmaxf(mn_in + var_in, 0.0f)); }
        return;
    }
    int off = offsets[c];
    float invm = 1.0f / (float)m;
    for (int d = tid; d < DF; d += 256) {
        float acc = 0.0f;
        for (int i = 0; i < m; ++i) acc += features[(size_t)member[off + i] * DF + d];
        meanb[d] = acc * invm;
    }
    __syncthreads();
    float p = 0.0f;
    for (int d = tid; d < DF; d += 256) { float a = fabsf(meanb[d]); float a2 = a * a; p += a2 * a2 * a; }
    red[tid] = p; __syncthreads();
    for (int o = 128; o >= 1; o >>= 1) { if (tid < o) red[tid] += red[tid + o]; __syncthreads(); }
    float mean_norm = powf(red[0], 0.4f);   // (sum|.|^5)^(2/5)
    __syncthreads();
    float varsum = 0.0f;
    for (int i = 0; i < m; ++i) {
        const float* f = features + (size_t)member[off + i] * DF;
        float part = 0.0f;
        for (int d = tid; d < DF; d += 256) {
            float a = fabsf(f[d] - meanb[d]); float a2 = a * a; part += a2 * a2 * a;
        }
        red[tid] = part; __syncthreads();
        for (int o = 128; o >= 1; o >>= 1) { if (tid < o) red[tid] += red[tid + o]; __syncthreads(); }
        varsum += powf(red[0], 0.4f);
        __syncthreads();
    }
    if (tid == 0) {
        float variance = varsum * invm;
        bool init = isnan(mn_in) || (mn_in == 0.0f);   // init keyed off mean_norm_ema only
        float mnf = init ? mean_norm : EMA_D * mn_in + (1.0f - EMA_D) * mean_norm;
        float vf  = init ? variance  : EMA_D * var_in + (1.0f - EMA_D) * variance;
        var_fin[c] = vf;
        s_arr[c] = cbrtf(fmaxf(mnf + vf, 0.0f));
    }
}

__global__ __launch_bounds__(1024) void reduce_s_kernel(const float* __restrict__ s_arr,
                                                        const float* __restrict__ var_fin,
                                                        float* scalars) {
    __shared__ float red[1024];
    int tid = threadIdx.x;
    red[tid] = (tid < NCLS) ? s_arr[tid] : 0.0f; __syncthreads();
    for (int o = 512; o >= 1; o >>= 1) { if (tid < o) red[tid] += red[tid + o]; __syncthreads(); }
    if (tid == 0) scalars[0] = red[0];
    __syncthreads();
    red[tid] = (tid < NCLS) ? var_fin[tid] : 0.0f; __syncthreads();
    for (int o = 512; o >= 1; o >>= 1) { if (tid < o) red[tid] += red[tid + o]; __syncthreads(); }
    if (tid == 0) scalars[1] = red[0] / (float)NCLS;   // bar_s
}

// cross-entropy with per-class margin gamma; one block per row
__global__ __launch_bounds__(256) void ce_kernel(const float* __restrict__ logits,
                                                 const int* __restrict__ targets,
                                                 const float* __restrict__ s_arr,
                                                 const float* __restrict__ scalars,
                                                 float* ce_acc) {
    __shared__ float red[256];
    int b = blockIdx.x, tid = threadIdx.x;
    int t = targets[b];
    float gamma = (float)NCLS * s_arr[t] / scalars[0];
    float inv = 1.0f / gamma;
    const float* row = logits + (size_t)b * NCLS;
    float mx = -INFINITY;
    for (int c = tid; c < NCLS; c += 256) mx = fmaxf(mx, row[c] * inv);
    red[tid] = mx; __syncthreads();
    for (int o = 128; o >= 1; o >>= 1) { if (tid < o) red[tid] = fmaxf(red[tid], red[tid + o]); __syncthreads(); }
    float M = red[0]; __syncthreads();
    float se = 0.0f;
    for (int c = tid; c < NCLS; c += 256) se += expf(row[c] * inv - M);
    red[tid] = se; __syncthreads();
    for (int o = 128; o >= 1; o >>= 1) { if (tid < o) red[tid] += red[tid + o]; __syncthreads(); }
    if (tid == 0) {
        float lp = row[t] * inv - M - logf(red[0]);
        atomicAdd(ce_acc, -lp);
    }
}

// rep loss: one wave per class; per-class Gram tiles via v_wmma_f32_16x16x32_f16
__global__ __launch_bounds__(32) void rep_kernel(
    const float* __restrict__ features, const int* __restrict__ counts,
    const int* __restrict__ offsets, const int* __restrict__ member,
    const float* __restrict__ sq, const float* __restrict__ scalars,
    float* rep_acc) {
    __shared__ _Float16 Ab[16 * AS];
    __shared__ _Float16 Bb[16 * AS];
    __shared__ float rs[16];
    const int c = blockIdx.x;
    const int m = counts[c];
    if (m < 2) return;                       // classes with <=1 sample contribute 0
    const int off = offsets[c];
    const float bar_s = scalars[1];
    const float inv10D = 1.0f / (10.0f * (float)DF);
    const int l = threadIdx.x;
    const int half = l >> 4, sub = l & 15;
    const int ntile = (m + 15) >> 4;
    float cls_sum = 0.0f;

    for (int ti = 0; ti < ntile; ++ti) {
        for (int r = 0; r < 16; ++r) {       // gather i-tile, f32 -> f16
            int gi = ti * 16 + r;
            if (gi < m) {
                const float* src = features + (size_t)member[off + gi] * DF;
                for (int e = l; e < DF; e += 32) Ab[r * AS + e] = (_Float16)src[e];
            } else {
                for (int e = l; e < DF; e += 32) Ab[r * AS + e] = (_Float16)0.0f;
            }
        }
        if (l < 16) rs[l] = 0.0f;
        __syncthreads();

        for (int tj = 0; tj < ntile; ++tj) {
            for (int r = 0; r < 16; ++r) {   // gather j-tile
                int gj = tj * 16 + r;
                if (gj < m) {
                    const float* src = features + (size_t)member[off + gj] * DF;
                    for (int e = l; e < DF; e += 32) Bb[r * AS + e] = (_Float16)src[e];
                } else {
                    for (int e = l; e < DF; e += 32) Bb[r * AS + e] = (_Float16)0.0f;
                }
            }
            __syncthreads();

            v8f acc = {};
            #pragma unroll
            for (int kk = 0; kk < DF; kk += 32) {   // D = F_i (16x32k) x F_j^T
                const int k0 = kk + half * 8;
                const int k1 = kk + 16 + half * 8;
                v8h alo = *(const v8h*)&Ab[sub * AS + k0];
                v8h ahi = *(const v8h*)&Ab[sub * AS + k1];
                v8h blo = *(const v8h*)&Bb[sub * AS + k0];
                v8h bhi = *(const v8h*)&Bb[sub * AS + k1];
                v16h a, b;
                #pragma unroll
                for (int i = 0; i < 8; ++i) {
                    a[i] = alo[i]; a[8 + i] = ahi[i];
                    b[i] = blo[i]; b[8 + i] = bhi[i];
                }
                acc = __builtin_amdgcn_wmma_f32_16x16x32_f16(
                    false, a, false, b, (short)0, acc, false, false);
            }

            #pragma unroll
            for (int r = 0; r < 8; ++r) {    // C/D layout: M = r + 8*half, N = sub
                int M = r + half * 8;
                int gi = ti * 16 + M;
                int gj = tj * 16 + sub;
                if (gi < m && gj < m && gi != gj) {
                    float d2 = sq[member[off + gi]] + sq[member[off + gj]] - 2.0f * acc[r];
                    float term = expf((d2 - 2.0f * bar_s) * inv10D);
                    atomicAdd(&rs[M], term);
                }
            }
            __syncthreads();
        }
        float part = 0.0f;
        if (l < 16 && (ti * 16 + l) < m) part = log1pf(rs[l]);
        #pragma unroll
        for (int o = 16; o >= 1; o >>= 1) part += __shfl_down(part, o, 32);
        if (l == 0) cls_sum += part;
        __syncthreads();
    }
    if (l == 0) atomicAdd(rep_acc, cls_sum / (float)m);   // per-class mean of log1p rows
}

__global__ void finish_kernel(const float* scalars, float* out, float invB) {
    out[0] = scalars[2] * invB +
             LAMBDA_R * scalars[3] * ((float)DF / 10.0f / (float)NCLS);
}

extern "C" void kernel_launch(void* const* d_in, const int* in_sizes, int n_in,
                              void* d_out, int out_size, void* d_ws, size_t ws_size,
                              hipStream_t stream) {
    const float* logits   = (const float*)d_in[0];
    const float* features = (const float*)d_in[1];
    const float* mn_ema   = (const float*)d_in[2];
    const float* var_ema  = (const float*)d_in[3];
    const int*   targets  = (const int*)d_in[4];
    float* out = (float*)d_out;
    const int B = in_sizes[4];

    float* ws = (float*)d_ws;
    float* sq      = ws;
    int*   counts  = (int*)(ws + 4096);
    int*   cursor  = counts + NCLS_PAD;
    int*   offsets = cursor + NCLS_PAD;
    int*   member  = offsets + NCLS_PAD;
    float* s_arr   = (float*)(member + 4096);
    float* var_fin = s_arr + NCLS_PAD;
    float* scalars = var_fin + NCLS_PAD;

    zero_kernel<<<(NCLS_PAD + 255) / 256, 256, 0, stream>>>(counts, cursor, scalars);
    hist_kernel<<<(B + 255) / 256, 256, 0, stream>>>(targets, counts, B);
    scan_kernel<<<1, NCLS_PAD, 0, stream>>>(counts, offsets);
    scatter_kernel<<<(B + 255) / 256, 256, 0, stream>>>(targets, offsets, cursor, member, B);
    sq_kernel<<<(B + 7) / 8, 256, 0, stream>>>(features, sq, B);
    cls_stats_kernel<<<NCLS, 256, 0, stream>>>(features, mn_ema, var_ema, counts, offsets,
                                               member, s_arr, var_fin);
    reduce_s_kernel<<<1, 1024, 0, stream>>>(s_arr, var_fin, scalars);
    ce_kernel<<<B, 256, 0, stream>>>(logits, targets, s_arr, scalars, scalars + 2);
    rep_kernel<<<NCLS, 32, 0, stream>>>(features, counts, offsets, member, sq, scalars,
                                        scalars + 3);
    finish_kernel<<<1, 1, 0, stream>>>(scalars, out, 1.0f / (float)B);
}
